// RegionProposalNetwork_61821759259236
// MI455X (gfx1250) — matrix operand order repaired and checked by reading, not compiled, not run on hardware
//
#include <hip/hip_runtime.h>
#include <math.h>

// ---------------- types for WMMA ----------------
typedef __attribute__((ext_vector_type(16))) __bf16        v16bf;
typedef __attribute__((ext_vector_type(8)))  float         v8f;
typedef __attribute__((ext_vector_type(8)))  unsigned short v8u16;
typedef __attribute__((ext_vector_type(16))) unsigned short v16u16;

// ---------------- problem constants ----------------
#define CCH   512
#define HH    64
#define WWID  64
#define NPIX  (HH*WWID)          // 4096
#define NA    9
#define NANCH (NPIX*NA)          // 36864
#define NSORT 65536
#define PRE_NMS  6000
#define POST_NMS 300
#define NMS_T    0.7f
#define MIN_SIZE 16.0f
#define BBOX_CLIP 4.135166556742356f   // log(1000/16)
#define NBLK  ((PRE_NMS + 31) / 32)    // 188

// ---------------- workspace layout (bytes) ----------------
static constexpr size_t WB_PLANE = (size_t)9 * CCH * CCH;            // u16 elems per plane
static constexpr size_t XB_PLANE = (size_t)66 * 66 * CCH;            // u16 elems per plane
static constexpr size_t OFF_WB   = 0;
static constexpr size_t SZ_WB    = 2 * WB_PLANE * sizeof(unsigned short);
static constexpr size_t OFF_XB   = OFF_WB + SZ_WB;
static constexpr size_t SZ_XB    = 2 * XB_PLANE * sizeof(unsigned short);
static constexpr size_t OFF_YT   = OFF_XB + SZ_XB;
static constexpr size_t SZ_YT    = (size_t)NPIX * CCH * sizeof(float);
static constexpr size_t OFF_SCORE= OFF_YT + SZ_YT;
static constexpr size_t SZ_SCORE = (size_t)NA * NPIX * sizeof(float);
static constexpr size_t OFF_BOXF = OFF_SCORE + SZ_SCORE;
static constexpr size_t SZ_BOXF  = (size_t)NA * 4 * NPIX * sizeof(float);
static constexpr size_t OFF_PRED = OFF_BOXF + SZ_BOXF;
static constexpr size_t SZ_PRED  = (size_t)NANCH * 4 * sizeof(float);
static constexpr size_t OFF_KEYS = OFF_PRED + SZ_PRED;
static constexpr size_t SZ_KEYS  = (size_t)NSORT * sizeof(unsigned long long);
static constexpr size_t OFF_BCLP = OFF_KEYS + SZ_KEYS;
static constexpr size_t SZ_BCLP  = (size_t)PRE_NMS * 4 * sizeof(float);
static constexpr size_t OFF_TOPV = OFF_BCLP + SZ_BCLP;
static constexpr size_t SZ_TOPV  = (size_t)PRE_NMS * sizeof(float);
static constexpr size_t OFF_VAL  = OFF_TOPV + SZ_TOPV;
static constexpr size_t SZ_VAL   = (size_t)PRE_NMS * sizeof(unsigned);
static constexpr size_t OFF_MASK = OFF_VAL + SZ_VAL;
static constexpr size_t SZ_MASK  = (size_t)PRE_NMS * NBLK * sizeof(unsigned);

// ---------------- helpers ----------------
__device__ __forceinline__ unsigned short f2bf_rne(float f) {
    unsigned u = __float_as_uint(f);
    unsigned r = u + 0x7FFFu + ((u >> 16) & 1u);
    return (unsigned short)(r >> 16);
}
__device__ __forceinline__ float bf2f(unsigned short h) {
    return __uint_as_float(((unsigned)h) << 16);
}

// ---------------- repack weights: conv1_w[co][ci][ky][kx] -> Wb[hi/lo][t][co][ci] ----------------
__global__ __launch_bounds__(256) void repack_w(const float* __restrict__ w,
                                                unsigned short* __restrict__ Wb) {
    int n = blockIdx.x * 256 + threadIdx.x;          // 9*512*512
    if (n >= 9 * CCH * CCH) return;
    int t  = n / (CCH * CCH);
    int r  = n % (CCH * CCH);
    int co = r / CCH;
    int ci = r % CCH;
    int ky = t / 3, kx = t % 3;
    float v = w[(((size_t)co * CCH + ci) * 3 + ky) * 3 + kx];
    unsigned short hi = f2bf_rne(v);
    float lo = v - bf2f(hi);
    size_t o = ((size_t)t * CCH + co) * CCH + ci;
    Wb[o]            = hi;
    Wb[o + WB_PLANE] = f2bf_rne(lo);
}

// ---------------- repack feature: X[ci][y][x] -> Xb[hi/lo][py][px][ci], zero padded ----------------
__global__ __launch_bounds__(256) void repack_x(const float* __restrict__ x,
                                                unsigned short* __restrict__ Xb) {
    int n = blockIdx.x * 256 + threadIdx.x;          // 66*66*512
    if (n >= 66 * 66 * CCH) return;
    int py = n / (66 * CCH);
    int r  = n % (66 * CCH);
    int px = r / CCH;
    int ci = r % CCH;
    float v = 0.f;
    if (py >= 1 && py <= HH && px >= 1 && px <= WWID)
        v = x[((size_t)ci * HH + (py - 1)) * WWID + (px - 1)];
    unsigned short hi = f2bf_rne(v);
    float lo = v - bf2f(hi);
    size_t o = (size_t)n;
    Xb[o]            = hi;
    Xb[o + XB_PLANE] = f2bf_rne(lo);
}

// ---------------- 3x3 conv via implicit GEMM + WMMA bf16 split ----------------
// grid (256 pixel-strips, 4 co-superblocks), block 256 (8 waves)
__global__ __launch_bounds__(256) void conv3_wmma(const unsigned short* __restrict__ Wb,
                                                  const unsigned short* __restrict__ Xb,
                                                  const float* __restrict__ bias,
                                                  float* __restrict__ yT) {
    int wave = threadIdx.x >> 5;
    int lane = threadIdx.x & 31;
    int hi16 = lane >> 4;        // 0 or 1
    int l    = lane & 15;
    int strip = blockIdx.x;                // 0..255
    int y  = strip >> 2;
    int x0 = (strip & 3) << 4;
    int co0 = blockIdx.y * 128 + wave * 16;
    int coA = co0 + l;                     // A-matrix row for this lane

    v8f acc = {};
    for (int t = 0; t < 9; ++t) {
        int ky = t / 3, kx = t % 3;
        const unsigned short* wb = Wb + ((size_t)(t * CCH + coA)) * CCH;
        const unsigned short* xb = Xb + ((size_t)(y + ky) * 66 + (x0 + l + kx)) * CCH;
        if (t < 8) {
            // pull next tap's panels toward WGP$ while this tap's WMMAs drain
            __builtin_prefetch(wb + (size_t)CCH * CCH, 0, 0);      // next A hi panel
            __builtin_prefetch(xb + ((t % 3 == 2) ? (size_t)(66 - 2) * CCH : (size_t)CCH), 0, 0);
        }
        for (int k0 = 0; k0 < CCH; k0 += 32) {
            // A 16x32 bf16 layout: lane<16 holds K {0..7,16..23}, lane>=16 holds {8..15,24..31}
            v8u16 ah0 = *(const v8u16*)(wb + k0 + hi16 * 8);
            v8u16 ah1 = *(const v8u16*)(wb + k0 + hi16 * 8 + 16);
            v8u16 al0 = *(const v8u16*)(wb + WB_PLANE + k0 + hi16 * 8);
            v8u16 al1 = *(const v8u16*)(wb + WB_PLANE + k0 + hi16 * 8 + 16);
            // B 32x16 bf16 layout: lane<16 holds K {0..7,16..23}, lane>=16 holds {8..15,24..31}
            v8u16 bh0 = *(const v8u16*)(xb + k0 + hi16 * 8);
            v8u16 bh1 = *(const v8u16*)(xb + k0 + hi16 * 8 + 16);
            v8u16 bl0 = *(const v8u16*)(xb + XB_PLANE + k0 + hi16 * 8);
            v8u16 bl1 = *(const v8u16*)(xb + XB_PLANE + k0 + hi16 * 8 + 16);
            v16u16 ahu, alu, bhu, blu;
#pragma unroll
            for (int i = 0; i < 8; ++i) {
                ahu[i] = ah0[i]; ahu[8 + i] = ah1[i];
                alu[i] = al0[i]; alu[8 + i] = al1[i];
                bhu[i] = bh0[i]; bhu[8 + i] = bh1[i];
                blu[i] = bl0[i]; blu[8 + i] = bl1[i];
            }
            v16bf aH = __builtin_bit_cast(v16bf, ahu);
            v16bf aL = __builtin_bit_cast(v16bf, alu);
            v16bf bH = __builtin_bit_cast(v16bf, bhu);
            v16bf bL = __builtin_bit_cast(v16bf, blu);
            acc = __builtin_amdgcn_wmma_f32_16x16x32_bf16(false, aH, false, bH, (short)0, acc, false, false);
            acc = __builtin_amdgcn_wmma_f32_16x16x32_bf16(false, aH, false, bL, (short)0, acc, false, false);
            acc = __builtin_amdgcn_wmma_f32_16x16x32_bf16(false, aL, false, bH, (short)0, acc, false, false);
        }
    }
    // D layout: VGPR r -> co = co0 + 8*hi16 + r ; N = l -> pixel
    int coD = co0 + hi16 * 8;
    int p   = y * WWID + x0 + l;
    float4 b0 = *(const float4*)(bias + coD);
    float4 b1 = *(const float4*)(bias + coD + 4);
    float4 o0, o1;
    o0.x = fmaxf(acc[0] + b0.x, 0.f); o0.y = fmaxf(acc[1] + b0.y, 0.f);
    o0.z = fmaxf(acc[2] + b0.z, 0.f); o0.w = fmaxf(acc[3] + b0.w, 0.f);
    o1.x = fmaxf(acc[4] + b1.x, 0.f); o1.y = fmaxf(acc[5] + b1.y, 0.f);
    o1.z = fmaxf(acc[6] + b1.z, 0.f); o1.w = fmaxf(acc[7] + b1.w, 0.f);
    float* out = yT + (size_t)p * CCH + coD;
    *(float4*)(out)     = o0;
    *(float4*)(out + 4) = o1;
}

// ---------------- 1x1 heads: scores (9) + box deltas (36) ----------------
__global__ __launch_bounds__(256) void heads(const float* __restrict__ yT,
                                             const float* __restrict__ clsw,
                                             const float* __restrict__ clsb,
                                             const float* __restrict__ boxw,
                                             const float* __restrict__ boxb,
                                             float* __restrict__ score_flat,
                                             float* __restrict__ box_flat) {
    __shared__ float ytile[16 * CCH];    // 32 KB
    int p0 = blockIdx.x * 16;
    const float4* src = (const float4*)(yT + (size_t)p0 * CCH);
    float4* dst = (float4*)ytile;
    for (int i = threadIdx.x; i < 16 * CCH / 4; i += 256) dst[i] = src[i];
    __syncthreads();
    for (int tt = threadIdx.x; tt < 45 * 16; tt += 256) {
        int o = tt >> 4, px = tt & 15;
        const float* wv;
        float b;
        if (o < NA) { wv = clsw + (size_t)o * CCH;        b = clsb[o]; }
        else        { wv = boxw + (size_t)(o - NA) * CCH; b = boxb[o - NA]; }
        const float4* y4 = (const float4*)(ytile + px * CCH);
        const float4* w4 = (const float4*)wv;
        float acc = 0.f;
        for (int k = 0; k < CCH / 4; ++k) {
            float4 a = y4[k], w = w4[k];
            acc += a.x * w.x + a.y * w.y + a.z * w.z + a.w * w.w;
        }
        float r = acc + b;
        if (o < NA) score_flat[(size_t)o * NPIX + p0 + px] = r;
        else        box_flat[(size_t)(o - NA) * NPIX + p0 + px] = r;
    }
}

// ---------------- decode + sigmoid + sort keys ----------------
__global__ __launch_bounds__(256) void decode(const float* __restrict__ score_flat,
                                              const float* __restrict__ box_flat,
                                              const float* __restrict__ anchors,
                                              float* __restrict__ preds,
                                              unsigned long long* __restrict__ keys) {
    int i = blockIdx.x * 256 + threadIdx.x;          // 0..65535
    if (i >= NSORT) return;
    if (i >= NANCH) { keys[i] = 0ull; return; }
    float sc = score_flat[i];
    float s  = 1.f / (1.f + expf(-sc));
    // deltas row i = box_flat[4i .. 4i+3] (flat (36,4096) order, as in the reference reshape)
    float dx = box_flat[4 * i + 0];
    float dy = box_flat[4 * i + 1];
    float dw = box_flat[4 * i + 2];
    float dh = box_flat[4 * i + 3];
    float a0 = anchors[4 * i + 0], a1 = anchors[4 * i + 1];
    float a2 = anchors[4 * i + 2], a3 = anchors[4 * i + 3];
    float aw = a2 - a0, ah = a3 - a1;
    float ax = a0 + 0.5f * aw, ay = a1 + 0.5f * ah;
    dw = fminf(dw, BBOX_CLIP);
    dh = fminf(dh, BBOX_CLIP);
    float px = dx * aw + ax;
    float py = dy * ah + ay;
    float pw = expf(dw) * aw;
    float ph = expf(dh) * ah;
    float4 pr;
    pr.x = px - 0.5f * pw; pr.y = py - 0.5f * ph;
    pr.z = px + 0.5f * pw; pr.w = py + 0.5f * ph;
    *(float4*)(preds + 4 * (size_t)i) = pr;
    // sigmoid in (0,1): float bits are order-preserving; ~i gives lower-index-first ties
    keys[i] = ((unsigned long long)__float_as_uint(s) << 32) | (unsigned)(0xFFFFFFFFu - (unsigned)i);
}

// ---------------- bitonic sort, descending; direction from GLOBAL index ----------------
// Global pass for j >= 4096 (cross-chunk exchanges)
__global__ __launch_bounds__(256) void bitonic_pass(unsigned long long* __restrict__ key,
                                                    int j, int k) {
    int i = blockIdx.x * 256 + threadIdx.x;
    int ixj = i ^ j;
    if (ixj <= i) return;
    unsigned long long a = key[i], b = key[ixj];
    bool descRegion = ((i & k) == 0);
    bool sw = descRegion ? (a < b) : (a > b);
    if (sw) { key[i] = b; key[ixj] = a; }
}

// Fused local stages: each block owns a 4096-key chunk in 32KB LDS.
// Covers all (k <= 4096, all j) — the opening 12 stages of the network.
__global__ __launch_bounds__(256) void bitonic_local_sort(unsigned long long* __restrict__ key) {
    __shared__ unsigned long long lk[4096];
    int base = blockIdx.x * 4096;
    for (int t = threadIdx.x; t < 4096; t += 256) lk[t] = key[base + t];
    __syncthreads();
    for (int k = 2; k <= 4096; k <<= 1) {
        for (int j = k >> 1; j > 0; j >>= 1) {
            for (int p = threadIdx.x; p < 2048; p += 256) {
                int i   = ((p & ~(j - 1)) << 1) | (p & (j - 1));   // bit j of i is 0
                int ixj = i | j;
                unsigned long long a = lk[i], b = lk[ixj];
                bool descRegion = (((base + i) & k) == 0);
                bool sw = descRegion ? (a < b) : (a > b);
                if (sw) { lk[i] = b; lk[ixj] = a; }
            }
            __syncthreads();
        }
    }
    for (int t = threadIdx.x; t < 4096; t += 256) key[base + t] = lk[t];
}

// Fused tail of a merge step k: all j = 2048..1 are chunk-local.
__global__ __launch_bounds__(256) void bitonic_local_merge(unsigned long long* __restrict__ key,
                                                           int k) {
    __shared__ unsigned long long lk[4096];
    int base = blockIdx.x * 4096;
    for (int t = threadIdx.x; t < 4096; t += 256) lk[t] = key[base + t];
    __syncthreads();
    for (int j = 2048; j > 0; j >>= 1) {
        for (int p = threadIdx.x; p < 2048; p += 256) {
            int i   = ((p & ~(j - 1)) << 1) | (p & (j - 1));
            int ixj = i | j;
            unsigned long long a = lk[i], b = lk[ixj];
            bool descRegion = (((base + i) & k) == 0);
            bool sw = descRegion ? (a < b) : (a > b);
            if (sw) { lk[i] = b; lk[ixj] = a; }
        }
        __syncthreads();
    }
    for (int t = threadIdx.x; t < 4096; t += 256) key[base + t] = lk[t];
}

// ---------------- gather top-6000: clip boxes, validity ----------------
__global__ __launch_bounds__(256) void gather_topk(const unsigned long long* __restrict__ keys,
                                                   const float* __restrict__ preds,
                                                   const int* __restrict__ ishape,
                                                   float* __restrict__ bclip,
                                                   float* __restrict__ topv,
                                                   unsigned* __restrict__ valid) {
    int j = blockIdx.x * 256 + threadIdx.x;
    if (j >= PRE_NMS) return;
    unsigned long long k = keys[j];
    float s = __uint_as_float((unsigned)(k >> 32));
    unsigned i = 0xFFFFFFFFu - (unsigned)(k & 0xFFFFFFFFull);
    float4 p = *(const float4*)(preds + 4 * (size_t)i);
    float wimg = (float)ishape[1], himg = (float)ishape[0];
    float4 b;
    b.x = fminf(fmaxf(p.x, 0.f), wimg);
    b.y = fminf(fmaxf(p.y, 0.f), himg);
    b.z = fminf(fmaxf(p.z, 0.f), wimg);
    b.w = fminf(fmaxf(p.w, 0.f), himg);
    *(float4*)(bclip + 4 * (size_t)j) = b;
    topv[j]  = s;
    valid[j] = ((b.z - b.x >= MIN_SIZE) && (b.w - b.y >= MIN_SIZE)) ? 1u : 0u;
}

// ---------------- NMS suppression bitmask: wave per (i, 32-wide j block) ----------------
__global__ __launch_bounds__(256) void nms_mask(const float* __restrict__ bclip,
                                                unsigned* __restrict__ mask) {
    int task = blockIdx.x * 8 + (threadIdx.x >> 5);
    int lane = threadIdx.x & 31;
    int i  = task / NBLK;
    int jb = task % NBLK;
    if (i >= PRE_NMS) return;
    if (jb < (i >> 5)) return;       // bits for j <= i are structurally 0 and never read
    int j = jb * 32 + lane;
    float4 bi = *(const float4*)(bclip + 4 * (size_t)i);
    float4 bj = (j < PRE_NMS) ? *(const float4*)(bclip + 4 * (size_t)j)
                              : float4{0.f, 0.f, 0.f, 0.f};
    float ai = (bi.z - bi.x) * (bi.w - bi.y);
    float aj = (bj.z - bj.x) * (bj.w - bj.y);
    float lx = fmaxf(bi.x, bj.x), ly = fmaxf(bi.y, bj.y);
    float rx = fminf(bi.z, bj.z), ry = fminf(bi.w, bj.w);
    float w = fmaxf(rx - lx, 0.f), h = fmaxf(ry - ly, 0.f);
    float inter = w * h;
    float uni = ai + aj - inter;
    float iou = (uni > 0.f) ? inter / uni : 0.f;
    bool pred = (j < PRE_NMS) && (j > i) && (iou > NMS_T);
    unsigned m = pred ? (1u << lane) : 0u;
#pragma unroll
    for (int off = 16; off > 0; off >>= 1) m |= __shfl_xor(m, off, 32);
    if (lane == 0) mask[(size_t)i * NBLK + jb] = m;
}

// ---------------- serial greedy NMS scan + final top-300 (single wave) ----------------
__global__ __launch_bounds__(32) void nms_serial(const unsigned* __restrict__ mask,
                                                 const unsigned* __restrict__ valid,
                                                 const float* __restrict__ bclip,
                                                 const float* __restrict__ topv,
                                                 float* __restrict__ out) {
    __shared__ unsigned remv[NBLK];
    __shared__ int sel[POST_NMS];
    int t = threadIdx.x;   // 0..31
    for (int w = t; w < NBLK; w += 32) {
        unsigned m = 0;
        for (int bit = 0; bit < 32; ++bit) {
            int j = w * 32 + bit;
            if (j >= PRE_NMS || !valid[j]) m |= (1u << bit);
        }
        remv[w] = m;
    }
    __syncthreads();
    for (int i = 0; i < PRE_NMS; ++i) {
        bool rem = (remv[i >> 5] >> (i & 31)) & 1u;
        if (!rem) {
            const unsigned* mrow = mask + (size_t)i * NBLK;
            int w0 = i >> 5;                       // only words >= i>>5 carry live bits
            for (int w = w0 + t; w < NBLK; w += 32) remv[w] |= mrow[w];
        }
        __syncthreads();   // single-wave workgroup: barrier lowers to S_NOP, acts as LDS fence
    }
    if (t == 0) {
        int cnt = 0;
        for (int j = 0; j < PRE_NMS && cnt < POST_NMS; ++j)
            if (!((remv[j >> 5] >> (j & 31)) & 1u)) sel[cnt++] = j;
        for (int j = 0; j < PRE_NMS && cnt < POST_NMS; ++j)
            if ((remv[j >> 5] >> (j & 31)) & 1u) sel[cnt++] = j;
    }
    __syncthreads();
    for (int k = t; k < POST_NMS; k += 32) {
        int j = sel[k];
        bool kept = !((remv[j >> 5] >> (j & 31)) & 1u);
        float4 b = *(const float4*)(bclip + 4 * (size_t)j);
        out[4 * k + 0] = b.x;
        out[4 * k + 1] = b.y;
        out[4 * k + 2] = b.z;
        out[4 * k + 3] = b.w;
        out[4 * POST_NMS + k] = kept ? topv[j] : -1.0f;
    }
}

// ---------------- launch ----------------
extern "C" void kernel_launch(void* const* d_in, const int* in_sizes, int n_in,
                              void* d_out, int out_size, void* d_ws, size_t ws_size,
                              hipStream_t stream) {
    const float* feat = (const float*)d_in[0];
    const float* anch = (const float*)d_in[1];
    const int*   ishp = (const int*)d_in[2];
    const float* w1   = (const float*)d_in[3];
    const float* b1   = (const float*)d_in[4];
    const float* clsw = (const float*)d_in[5];
    const float* clsb = (const float*)d_in[6];
    const float* boxw = (const float*)d_in[7];
    const float* boxb = (const float*)d_in[8];
    float* out = (float*)d_out;

    char* ws = (char*)d_ws;
    unsigned short* Wb        = (unsigned short*)(ws + OFF_WB);
    unsigned short* Xb        = (unsigned short*)(ws + OFF_XB);
    float* yT                 = (float*)(ws + OFF_YT);
    float* score_flat         = (float*)(ws + OFF_SCORE);
    float* box_flat           = (float*)(ws + OFF_BOXF);
    float* preds              = (float*)(ws + OFF_PRED);
    unsigned long long* keys  = (unsigned long long*)(ws + OFF_KEYS);
    float* bclip              = (float*)(ws + OFF_BCLP);
    float* topv               = (float*)(ws + OFF_TOPV);
    unsigned* valid           = (unsigned*)(ws + OFF_VAL);
    unsigned* mask            = (unsigned*)(ws + OFF_MASK);

    repack_w<<<(9 * CCH * CCH + 255) / 256, 256, 0, stream>>>(w1, Wb);
    repack_x<<<(66 * 66 * CCH + 255) / 256, 256, 0, stream>>>(feat, Xb);
    conv3_wmma<<<dim3(256, 4), 256, 0, stream>>>(Wb, Xb, b1, yT);
    heads<<<NPIX / 16, 256, 0, stream>>>(yT, clsw, clsb, boxw, boxb, score_flat, box_flat);
    decode<<<NSORT / 256, 256, 0, stream>>>(score_flat, box_flat, anch, preds, keys);

    // Bitonic sort, 15 launches total:
    //   1 fused local-sort (k=2..4096), then per k in {8192..65536}:
    //   global passes for j>=4096, one fused local merge for j<=2048.
    bitonic_local_sort<<<NSORT / 4096, 256, 0, stream>>>(keys);
    for (int k = 8192; k <= NSORT; k <<= 1) {
        for (int j = k >> 1; j >= 4096; j >>= 1)
            bitonic_pass<<<NSORT / 256, 256, 0, stream>>>(keys, j, k);
        bitonic_local_merge<<<NSORT / 4096, 256, 0, stream>>>(keys, k);
    }

    gather_topk<<<(PRE_NMS + 255) / 256, 256, 0, stream>>>(keys, preds, ishp, bclip, topv, valid);
    nms_mask<<<(PRE_NMS * NBLK) / 8, 256, 0, stream>>>(bclip, mask);
    nms_serial<<<1, 32, 0, stream>>>(mask, valid, bclip, topv, out);
}
